// MultiHeadAtt_75505525064174
// MI455X (gfx1250) — matrix operand order repaired and checked by reading, not compiled, not run on hardware
//
#include <hip/hip_runtime.h>
#include <hip/hip_bf16.h>
#include <math.h>

// ---------------------------------------------------------------------------
// Problem constants (match reference)
// ---------------------------------------------------------------------------
#define B_  2
#define L_  2048
#define D_  1024
#define H_  16
#define DK_ 64
#define BL_ (B_ * L_)          // 4096
#define EPS_ 1e-6f
#define NEG_BIG (-1.0e30f)

typedef __attribute__((ext_vector_type(16))) __bf16 v16bf;
typedef __attribute__((ext_vector_type(8)))  __bf16 v8bf;
typedef __attribute__((ext_vector_type(4)))  __bf16 v4bf;
typedef __attribute__((ext_vector_type(8)))  float  v8f;

// ---------------------------------------------------------------------------
// Fragment builders using contiguous k = 16*g + e for both A and B (not the ISA table order), arranged so
// every fragment is two contiguous 128-bit loads per lane.
//
// A (16x32 bf16, MxK): lane g=lane>>4, m=lane&15.
//   e=0..15 -> K = 16*g + e         (one contiguous run)
// B (32x16 bf16, KxN): n=lane&15, g=lane>>4; e -> K = 16*g + e
//   -> from B^T storage [N][K]: one contiguous 16-element run.
// C/D (16x16 f32): element r: M = r + 8*(lane>>4), N = lane&15
// ---------------------------------------------------------------------------
__device__ __forceinline__ v16bf cat8(v8bf lo, v8bf hi) {
    return __builtin_shufflevector(lo, hi, 0, 1, 2, 3, 4, 5, 6, 7,
                                   8, 9, 10, 11, 12, 13, 14, 15);
}

// src: row-major [M][K] with leading dim ld (multiple of 8, 16B-aligned base)
__device__ __forceinline__ v16bf load_a_frag(const __bf16* src, int ld) {
    const int lane = threadIdx.x & 31;
    const int m = lane & 15;
    const int g = lane >> 4;
    const __bf16* p = src + m * ld + 16 * g;
    v8bf lo = *(const v8bf*)(p);          // K = 16g .. 16g+7
    v8bf hi = *(const v8bf*)(p + 8);      // K = 16g+8 .. 16g+15
    return cat8(lo, hi);
}

// srcT: row-major B^T, i.e. [N][K] with leading dim ld (multiple of 8)
__device__ __forceinline__ v16bf load_b_fragT(const __bf16* srcT, int ld) {
    const int lane = threadIdx.x & 31;
    const int n = lane & 15;
    const int g = lane >> 4;
    const __bf16* p = srcT + n * ld + 16 * g;   // K = 16g .. 16g+15
    v8bf lo = *(const v8bf*)(p);
    v8bf hi = *(const v8bf*)(p + 8);
    return cat8(lo, hi);
}

// B-fragment of W^T taken directly from row-major f32 W[out][in]:
// B[k][n] = W[n][k] -> per lane 16 consecutive f32 -> 4x global_load_b128.
__device__ __forceinline__ v16bf load_b_frag_wT(const float* w, int n0, int k0, int ldw) {
    const int lane = threadIdx.x & 31;
    const int n = n0 + (lane & 15);
    const int g = lane >> 4;
    const float4* p = (const float4*)(w + (size_t)n * ldw + k0 + 16 * g);
    v16bf b;
#pragma unroll
    for (int j = 0; j < 4; ++j) {
        float4 x = p[j];
        b[4 * j + 0] = (__bf16)x.x;
        b[4 * j + 1] = (__bf16)x.y;
        b[4 * j + 2] = (__bf16)x.z;
        b[4 * j + 3] = (__bf16)x.w;
    }
    return b;
}

__device__ __forceinline__ v8f wmma_bf16(v16bf a, v16bf b, v8f c) {
    // D = A*B + C (f32 accumulate) -> v_wmma_f32_16x16x32_bf16
    return __builtin_amdgcn_wmma_f32_16x16x32_bf16(
        /*neg_a=*/false, a, /*neg_b=*/false, b,
        /*c_mod=*/(short)0, c, /*reuse_a=*/false, /*reuse_b=*/false);
}

// ---------------------------------------------------------------------------
// Kernel 1: per-head projections   Y[h] = X @ W[h]
//   X: [BL, D] f32    W: [H, D, DK] f32    Y: [H, BL, DK] bf16
// grid = (BL/32, H, 3), block = 256 (8 waves).
// wave w -> row tile (w>>2)*16 of 32 rows, col tile (w&3)*16 of 64 cols.
// ---------------------------------------------------------------------------
__global__ void __launch_bounds__(256)
mha_proj_kernel(const float* __restrict__ q, const float* __restrict__ k,
                const float* __restrict__ v,
                const float* __restrict__ w_qs, const float* __restrict__ w_ks,
                const float* __restrict__ w_vs,
                __bf16* __restrict__ qs, __bf16* __restrict__ ks,
                __bf16* __restrict__ vs) {
    const int tid  = threadIdx.x;
    const int wave = tid >> 5;
    const int lane = tid & 31;
    const int row0 = blockIdx.x * 32;
    const int h    = blockIdx.y;
    const int mat  = blockIdx.z;

    const float* X = (mat == 0) ? q : (mat == 1) ? k : v;
    const float* W = ((mat == 0) ? w_qs : (mat == 1) ? w_ks : w_vs)
                     + (size_t)h * D_ * DK_;
    __bf16* Y = ((mat == 0) ? qs : (mat == 1) ? ks : vs)
                + (size_t)h * BL_ * DK_;

    __shared__ __align__(16) __bf16 sA[32 * 32];    // [row][k]  row-major
    __shared__ __align__(16) __bf16 sBt[64 * 32];   // W^T: [n][k]

    const int rt = wave >> 2;        // 0..1
    const int ct = wave & 3;         // 0..3

    v8f acc = {};
    for (int k0 = 0; k0 < D_; k0 += 32) {
        __syncthreads();
        // stage A chunk: one float4 per thread, packed bf16 x4 store
        {
            int r = tid >> 3;               // 0..31
            int c = (tid & 7) * 4;          // 0..28
            float4 x = *(const float4*)&X[(size_t)(row0 + r) * D_ + k0 + c];
            v4bf y = { (__bf16)x.x, (__bf16)x.y, (__bf16)x.z, (__bf16)x.w };
            *(v4bf*)(sA + r * 32 + c) = y;
        }
        // stage W^T chunk: contiguous float4 reads along n, transposed stores
#pragma unroll
        for (int j = tid; j < 512; j += 256) {
            int kk = j >> 4;                // 0..31
            int nq = (j & 15) * 4;          // 0..60
            float4 x = *(const float4*)&W[(size_t)(k0 + kk) * DK_ + nq];
            sBt[(nq + 0) * 32 + kk] = (__bf16)x.x;
            sBt[(nq + 1) * 32 + kk] = (__bf16)x.y;
            sBt[(nq + 2) * 32 + kk] = (__bf16)x.z;
            sBt[(nq + 3) * 32 + kk] = (__bf16)x.w;
        }
        __syncthreads();
        v16bf a = load_a_frag(sA + rt * 16 * 32, 32);
        v16bf b = load_b_fragT(sBt + (ct * 16) * 32, 32);
        acc = wmma_bf16(a, b, acc);
    }

    // store C fragment (bf16)
    const int g = lane >> 4;
    const int n = lane & 15;
#pragma unroll
    for (int r = 0; r < 8; ++r) {
        int row = row0 + rt * 16 + r + 8 * g;
        int col = ct * 16 + n;
        Y[(size_t)row * DK_ + col] = (__bf16)acc[r];
    }
}

// ---------------------------------------------------------------------------
// Kernel 2: flash attention per (h, b).
// grid = (L/64, B, H), block = 128 (4 waves); wave handles 16 query rows.
// K chunk stays row-major [key][dk] (it IS B^T for S = Q*K^T).
// V chunk staged transposed [dk][key] (B^T for O = P*V).
// ---------------------------------------------------------------------------
__global__ void __launch_bounds__(128)
mha_attn_kernel(const __bf16* __restrict__ qs, const __bf16* __restrict__ ks,
                const __bf16* __restrict__ vs, __bf16* __restrict__ attn_cat) {
    const int tid  = threadIdx.x;
    const int wave = tid >> 5;
    const int lane = tid & 31;
    const int g    = lane >> 4;
    const int ln   = lane & 15;
    const int b    = blockIdx.y;
    const int h    = blockIdx.z;
    const int q0   = blockIdx.x * 64 + wave * 16;   // query row within [0,L)

    const __bf16* qs_hb = qs + ((size_t)h * BL_ + (size_t)b * L_) * DK_;
    const __bf16* ks_hb = ks + ((size_t)h * BL_ + (size_t)b * L_) * DK_;
    const __bf16* vs_hb = vs + ((size_t)h * BL_ + (size_t)b * L_) * DK_;

    __shared__ __align__(16) __bf16 sK[32 * DK_];      // [key][dk]  (= B^T for S)
    __shared__ __align__(16) __bf16 sVt[DK_ * 32];     // [dk][key]  (= B^T for O)
    __shared__ __align__(16) __bf16 sP[4 * 16 * 32];   // per-wave P [m][k], ld=32

    // Q fragments stay resident (A layout, K = dk)
    v16bf qa0 = load_a_frag(qs_hb + (size_t)q0 * DK_ + 0,  DK_);
    v16bf qa1 = load_a_frag(qs_hb + (size_t)q0 * DK_ + 32, DK_);

    v8f O[4] = {v8f{}, v8f{}, v8f{}, v8f{}};
    float m_i[8], l_i[8];
#pragma unroll
    for (int r = 0; r < 8; ++r) { m_i[r] = NEG_BIG; l_i[r] = 0.0f; }

    const float scale = 0.125f;            // 1/sqrt(64)

    for (int kc = 0; kc < L_ / 32; ++kc) {
        const int kst = kc * 32;
        __syncthreads();
        // stage K chunk: straight 16B-vector copy (256 x uint4)
        {
            const uint4* src = (const uint4*)(ks_hb + (size_t)kst * DK_);
            uint4* dst = (uint4*)sK;
#pragma unroll
            for (int i = tid; i < 256; i += 128) dst[i] = src[i];
        }
        // stage V chunk transposed: vector reads along dk, scattered b16 stores
#pragma unroll
        for (int j = tid; j < 256; j += 128) {
            int key = j >> 3;               // 0..31
            int dq  = (j & 7) * 8;          // 0..56
            v8bf x = *(const v8bf*)(vs_hb + (size_t)(kst + key) * DK_ + dq);
#pragma unroll
            for (int e = 0; e < 8; ++e) sVt[(dq + e) * 32 + key] = x[e];
        }
        if (kc + 1 < L_ / 32) {            // hint next chunk toward the caches
            __builtin_prefetch(ks_hb + (size_t)(kst + 32) * DK_ + lane * DK_, 0, 0);
            __builtin_prefetch(vs_hb + (size_t)(kst + 32) * DK_ + lane * DK_, 0, 0);
        }
        __syncthreads();

        // S = Q @ K^T  (16 q x 32 keys), two N-tiles of 16 keys.
        // B-frag tile t, dk-chunk c lives at sK + (t*16)*64 + c*32 (contiguous).
        v8f s0 = {}, s1 = {};
        s0 = wmma_bf16(qa0, load_b_fragT(sK + 0 * 16 * DK_ + 0,  DK_), s0);
        s0 = wmma_bf16(qa1, load_b_fragT(sK + 0 * 16 * DK_ + 32, DK_), s0);
        s1 = wmma_bf16(qa0, load_b_fragT(sK + 1 * 16 * DK_ + 0,  DK_), s1);
        s1 = wmma_bf16(qa1, load_b_fragT(sK + 1 * 16 * DK_ + 32, DK_), s1);

        __bf16* myP = sP + wave * (16 * 32);
#pragma unroll
        for (int r = 0; r < 8; ++r) {
            float a0 = s0[r] * scale;
            float a1 = s1[r] * scale;
            // row max over the 16 lanes of this half (deterministic butterfly)
            float vm = fmaxf(a0, a1);
#pragma unroll
            for (int msk = 1; msk <= 8; msk <<= 1)
                vm = fmaxf(vm, __shfl_xor(vm, msk, 32));
            float m_new = fmaxf(m_i[r], vm);
            float corr  = __expf(m_i[r] - m_new);
            float p0 = __expf(a0 - m_new);
            float p1 = __expf(a1 - m_new);
            float rs = p0 + p1;
#pragma unroll
            for (int msk = 1; msk <= 8; msk <<= 1)
                rs += __shfl_xor(rs, msk, 32);
            l_i[r] = l_i[r] * corr + rs;
            m_i[r] = m_new;
#pragma unroll
            for (int t = 0; t < 4; ++t) O[t][r] *= corr;
            int m = r + 8 * g;
            myP[m * 32 + ln]      = (__bf16)p0;
            myP[m * 32 + 16 + ln] = (__bf16)p1;
        }
        // same-wave LDS store->load; backend inserts s_wait_dscnt for the dep
        v16bf pa = load_a_frag(myP, 32);
#pragma unroll
        for (int t = 0; t < 4; ++t)
            O[t] = wmma_bf16(pa, load_b_fragT(sVt + (t * 16) * 32, 32), O[t]);
    }

    // normalize and store head-interleaved bf16: attn_cat[b, l, h*64 + dk]
#pragma unroll
    for (int r = 0; r < 8; ++r) {
        float inv = 1.0f / l_i[r];
        int row = b * L_ + q0 + r + 8 * g;
#pragma unroll
        for (int t = 0; t < 4; ++t) {
            int feat = h * DK_ + t * 16 + ln;
            attn_cat[(size_t)row * D_ + feat] = (__bf16)(O[t][r] * inv);
        }
    }
}

// ---------------------------------------------------------------------------
// Kernel 3: out = attn_cat @ w_o^T + q, then LayerNorm (unbiased std, eps).
// grid = BL/16, block = 256 (8 waves); block computes 16 rows x 1024 cols.
// wave w covers N in [w*128, w*128+128) -> 8 accumulators.
// ---------------------------------------------------------------------------
__global__ void __launch_bounds__(256)
mha_oproj_ln_kernel(const __bf16* __restrict__ attn_cat,
                    const float* __restrict__ w_o,
                    const float* __restrict__ q_res,
                    const float* __restrict__ gamma,
                    const float* __restrict__ beta,
                    float* __restrict__ out) {
    const int tid  = threadIdx.x;
    const int wave = tid >> 5;
    const int lane = tid & 31;
    const int g    = lane >> 4;
    const int ln   = lane & 15;
    const int m0   = blockIdx.x * 16;       // global row in [0, BL)
    const int nb   = wave * 128;

    __shared__ __align__(16) __bf16 sA[16 * 32];
    __shared__ float s_sum[8][16];
    __shared__ float s_sq[8][16];

    v8f acc[8] = {v8f{}, v8f{}, v8f{}, v8f{}, v8f{}, v8f{}, v8f{}, v8f{}};

    for (int k0 = 0; k0 < D_; k0 += 32) {
        __syncthreads();
        if (tid < 64) {                      // 64 x uint4 = 16x32 bf16
            int r = tid >> 2;
            int c = (tid & 3) * 8;
            *(uint4*)(sA + r * 32 + c) =
                *(const uint4*)(attn_cat + (size_t)(m0 + r) * D_ + k0 + c);
        }
        __syncthreads();
        v16bf a = load_a_frag(sA, 32);
#pragma unroll
        for (int t = 0; t < 8; ++t) {
            v16bf bfrag = load_b_frag_wT(w_o, nb + t * 16, k0, D_);
            acc[t] = wmma_bf16(a, bfrag, acc[t]);
        }
    }

    // residual + deterministic per-row sum / sumsq reduction
#pragma unroll
    for (int r = 0; r < 8; ++r) {
        int m   = r + 8 * g;
        int row = m0 + m;
        float ps = 0.0f, pq = 0.0f;
#pragma unroll
        for (int t = 0; t < 8; ++t) {
            int n = nb + t * 16 + ln;
            float y = acc[t][r] + q_res[(size_t)row * D_ + n];
            acc[t][r] = y;
            ps += y;
            pq += y * y;
        }
#pragma unroll
        for (int msk = 1; msk <= 8; msk <<= 1) {
            ps += __shfl_xor(ps, msk, 32);
            pq += __shfl_xor(pq, msk, 32);
        }
        if (ln == 0) { s_sum[wave][m] = ps; s_sq[wave][m] = pq; }
    }
    __syncthreads();

#pragma unroll
    for (int r = 0; r < 8; ++r) {
        int m   = r + 8 * g;
        int row = m0 + m;
        float ts = 0.0f, tq = 0.0f;
#pragma unroll
        for (int w = 0; w < 8; ++w) { ts += s_sum[w][m]; tq += s_sq[w][m]; }
        float mean = ts * (1.0f / (float)D_);
        float var  = (tq - (float)D_ * mean * mean) * (1.0f / (float)(D_ - 1));
        var = fmaxf(var, 0.0f);
        float sig = fmaxf(sqrtf(var), EPS_);
        float inv = 1.0f / sig;
#pragma unroll
        for (int t = 0; t < 8; ++t) {
            int n = nb + t * 16 + ln;
            out[(size_t)row * D_ + n] =
                (acc[t][r] - mean) * inv * gamma[n] + beta[n];
        }
    }
}

// ---------------------------------------------------------------------------
// Host launcher
// ---------------------------------------------------------------------------
extern "C" void kernel_launch(void* const* d_in, const int* in_sizes, int n_in,
                              void* d_out, int out_size, void* d_ws, size_t ws_size,
                              hipStream_t stream) {
    const float* q     = (const float*)d_in[0];
    const float* k     = (const float*)d_in[1];
    const float* v     = (const float*)d_in[2];
    const float* w_qs  = (const float*)d_in[3];
    const float* w_ks  = (const float*)d_in[4];
    const float* w_vs  = (const float*)d_in[5];
    const float* w_o   = (const float*)d_in[6];
    const float* gamma = (const float*)d_in[7];
    const float* beta  = (const float*)d_in[8];
    float* out = (float*)d_out;

    const size_t proj_bytes = (size_t)H_ * BL_ * DK_ * sizeof(__bf16); // 8 MB each
    char* ws = (char*)d_ws;
    __bf16* qs       = (__bf16*)(ws + 0 * proj_bytes);
    __bf16* ks       = (__bf16*)(ws + 1 * proj_bytes);
    __bf16* vs       = (__bf16*)(ws + 2 * proj_bytes);
    __bf16* attn_cat = (__bf16*)(ws + 3 * proj_bytes);                 // BL*D bf16 = 8 MB

    // 1) per-head projections
    mha_proj_kernel<<<dim3(BL_ / 32, H_, 3), 256, 0, stream>>>(
        q, k, v, w_qs, w_ks, w_vs, qs, ks, vs);

    // 2) flash attention
    mha_attn_kernel<<<dim3(L_ / 64, B_, H_), 128, 0, stream>>>(
        qs, ks, vs, attn_cat);

    // 3) output projection + residual + LayerNorm
    mha_oproj_ln_kernel<<<dim3(BL_ / 16), 256, 0, stream>>>(
        attn_cat, w_o, q, gamma, beta, out);
}
